// Actor_80994493268285
// MI455X (gfx1250) — matrix operand order repaired and checked by reading, not compiled, not run on hardware
//
#include <hip/hip_runtime.h>

typedef __attribute__((ext_vector_type(16))) _Float16 v16h;
typedef __attribute__((ext_vector_type(8)))  _Float16 v8h;
typedef __attribute__((ext_vector_type(2)))  __fp16   hp2;   // cvt_pkrtz result type
typedef __attribute__((ext_vector_type(8)))  float    v8f;
typedef __attribute__((ext_vector_type(4)))  float    f4;

// ---------------------------------------------------------------------------
// WMMA helper (CDNA5 V_WMMA_F32_16X16X32_F16, wave32)
// ---------------------------------------------------------------------------
__device__ __forceinline__ v8f wmma_f16(v16h a, v16h b, v8f c) {
  return __builtin_amdgcn_wmma_f32_16x16x32_f16(false, a, false, b, (short)0, c,
                                                false, false);
}

// A fragment: 16x32 (MxK) f16 from row-major LDS (leading dim ld).
// Per lane: 16 contiguous halves (32B): halves 0..15 = K[kB + 16*hi .. +15]
__device__ __forceinline__ v16h load_a_frag(const _Float16* A, int ld, int mB,
                                            int kB, int lane) {
  const int m  = mB + (lane & 15);
  const int hi = (lane >> 4) & 1;
  const _Float16* row = A + m * ld;
  return *(const v16h*)(row + kB + 16 * hi);
}

// B fragment where B[k][n] = W[n][k] (row-major [N][K] storage, i.e. weight
// [out][in] or a transposed activation). Per-lane: 16 contiguous halves (32B).
__device__ __forceinline__ v16h load_bT_frag(const _Float16* W, int ld, int nB,
                                             int kB, int lane) {
  const int n  = nB + (lane & 15);
  const int hi = (lane >> 4) & 1;
  return *(const v16h*)(W + n * ld + kB + 16 * hi);
}

// D store: 16x16 f32 acc -> f16 LDS (+bias, optional relu).
// Optionally also writes the transposed tile (contiguous 8-half store).
__device__ __forceinline__ void store_tile(v8f acc, _Float16* O, _Float16* OT,
                                           int mB, int nB, int lane,
                                           const float* bias, bool relu) {
  const int n  = nB + (lane & 15);
  const int hi = (lane >> 4) & 1;
  const float bv = bias ? bias[n] : 0.0f;
  v8h packed;
#pragma unroll
  for (int v = 0; v < 8; ++v) {
    float x = acc[v] + bv;
    if (relu) x = fmaxf(x, 0.0f);
    const _Float16 hx = (_Float16)x;
    O[(mB + v + 8 * hi) * 64 + n] = hx;
    packed[v] = hx;
  }
  if (OT) *(v8h*)(OT + n * 64 + mB + 8 * hi) = packed;  // 16B-aligned
}

// 64x64-output GEMM: O = act(A[64xK] @ B^T + bias); B row-major [64][K].
// 8 waves -> 2 tiles/wave.
__device__ __forceinline__ void gemm64(const _Float16* A, const _Float16* Bt,
                                       int K, const float* bias, bool relu,
                                       _Float16* O, _Float16* OT, int wave,
                                       int lane) {
#pragma unroll
  for (int t = 0; t < 2; ++t) {
    const int tile = wave * 2 + t;
    const int mB = (tile >> 2) * 16, nB = (tile & 3) * 16;
    v8f acc = {};
    for (int k = 0; k < K; k += 32) {
      acc = wmma_f16(load_a_frag(A, K, mB, k, lane),
                     load_bT_frag(Bt, K, nB, k, lane), acc);
    }
    store_tile(acc, O, OT, mB, nB, lane, bias, relu);
  }
}

__device__ __forceinline__ float sigmoidf_(float x) {
  return 1.0f / (1.0f + __expf(-x));
}
__device__ __forceinline__ float tanhf_(float x) {
  const float t = __expf(-2.0f * x);
  return (1.0f - t) / (1.0f + t);
}

// Vectorized f32 -> f16 panel staging (count % 1024 == 0). NT variant keeps
// read-once streams (state/adj) from churning L2; temporal variant is used for
// data that is re-read (weights each block, hidden's f32 blend read).
template <bool NT>
__device__ __forceinline__ void cvt_panel(const float* __restrict__ src,
                                          _Float16* __restrict__ dst, int count,
                                          int tid) {
  const f4* s4 = (const f4*)src;
  for (int i = tid; i < (count >> 2); i += 256) {
    const f4 v = NT ? __builtin_nontemporal_load(&s4[i]) : s4[i];
    const hp2 p0 = __builtin_amdgcn_cvt_pkrtz(v[0], v[1]);
    const hp2 p1 = __builtin_amdgcn_cvt_pkrtz(v[2], v[3]);
    *(hp2*)(dst + i * 4)     = p0;
    *(hp2*)(dst + i * 4 + 2) = p1;
  }
}

// ---------------------------------------------------------------------------
// One workgroup per batch element b. 256 threads = 8 wave32.
// ---------------------------------------------------------------------------
__global__ __launch_bounds__(256, 1) void actor_fused_kernel(
    const float* __restrict__ state, const float* __restrict__ adjc,
    const float* __restrict__ adjo, const float* __restrict__ hidden,
    const float* __restrict__ fc1w, const float* __restrict__ fc1b,
    const float* __restrict__ fc2w, const float* __restrict__ fc2b,
    const float* __restrict__ compw, const float* __restrict__ compb,
    const float* __restrict__ coopw, const float* __restrict__ coopb,
    const float* __restrict__ fc3w, const float* __restrict__ fc3b,
    const float* __restrict__ outw, const float* __restrict__ outb,
    const float* __restrict__ wih, const float* __restrict__ whh,
    const float* __restrict__ bih, const float* __restrict__ bhh,
    float* __restrict__ out) {
  // Weights (f16)
  __shared__ _Float16 sW1[64 * 64], sW2[64 * 64], sWc[64 * 64], sWo[64 * 64];
  __shared__ _Float16 sW3[64 * 192];  // [out=64][in=192]
  __shared__ _Float16 sWih[192 * 64], sWhh[192 * 64];
  // Biases (f32)
  __shared__ float sB1[64], sB2[64], sBc[64], sBo[64], sB3[64], sOw[64];
  __shared__ float sBih[192], sBhh[192];
  // Activation buffers (f16, 64x64), reused across stages
  __shared__ _Float16 buf0[64 * 64], buf1[64 * 64], buf2[64 * 64];
  __shared__ _Float16 buf3[64 * 64], buf4[64 * 64], buf5[64 * 64];

  const int tid  = threadIdx.x;
  const int lane = tid & 31;
  const int wave = tid >> 5;
  const int b    = blockIdx.x;
  const size_t ob = (size_t)b * 4096;

  // ---- stage weights (L2-resident, temporal loads) ----
  cvt_panel<false>(fc1w, sW1, 4096, tid);
  cvt_panel<false>(fc2w, sW2, 4096, tid);
  cvt_panel<false>(compw, sWc, 4096, tid);
  cvt_panel<false>(coopw, sWo, 4096, tid);
  cvt_panel<false>(fc3w, sW3, 64 * 192, tid);
  cvt_panel<false>(wih, sWih, 192 * 64, tid);
  cvt_panel<false>(whh, sWhh, 192 * 64, tid);
  if (tid < 64) {
    sB1[tid] = fc1b[tid]; sB2[tid] = fc2b[tid]; sBc[tid] = compb[tid];
    sBo[tid] = coopb[tid]; sB3[tid] = fc3b[tid]; sOw[tid] = outw[tid];
  }
  if (tid < 192) { sBih[tid] = bih[tid]; sBhh[tid] = bhh[tid]; }
  // ---- stage per-batch panels ----
  // state/adj: read exactly once -> non-temporal stream.
  // hidden: re-read as f32 in the GRU blend -> temporal, so the 16KB panel is
  // still L2-resident for the second read (saves ~64MB of HBM traffic).
  cvt_panel<true>(state + ob, buf0, 4096, tid);
  cvt_panel<true>(adjc + ob, buf1, 4096, tid);
  cvt_panel<true>(adjo + ob, buf2, 4096, tid);
  cvt_panel<false>(hidden + ob, buf3, 4096, tid);
  __syncthreads();

  // 1) x1 = relu(state @ fc1^T + b1)                buf0 -> buf4
  gemm64(buf0, sW1, 64, sB1, true, buf4, nullptr, wave, lane);
  __syncthreads();
  // 2) x2 = relu(x1 @ fc2^T + b2)                   buf4 -> buf5 (+x2^T -> buf0)
  gemm64(buf4, sW2, 64, sB2, true, buf5, buf0, wave, lane);
  __syncthreads();
  // 3) aggC = adj_comp @ x2   (B from x2^T)         buf1,buf0 -> buf4
  gemm64(buf1, buf0, 64, nullptr, false, buf4, nullptr, wave, lane);
  __syncthreads();
  // 4) aggO = adj_coop @ x2   (B from x2^T)         buf2,buf0 -> buf1
  gemm64(buf2, buf0, 64, nullptr, false, buf1, nullptr, wave, lane);
  __syncthreads();
  // 5) xcomp = relu(aggC @ comp^T + bc)             buf4 -> buf2
  gemm64(buf4, sWc, 64, sBc, true, buf2, nullptr, wave, lane);
  // 6) xcoop = relu(aggO @ coop^T + bo)             buf1 -> buf0
  gemm64(buf1, sWo, 64, sBo, true, buf0, nullptr, wave, lane);
  __syncthreads();

  // 7) x3 = relu([x2|xcomp|xcoop] @ fc3^T + b3), K=192 virtual concat -> buf4
  {
    const _Float16* srcs[3] = {buf5, buf2, buf0};
#pragma unroll
    for (int t = 0; t < 2; ++t) {
      const int tile = wave * 2 + t;
      const int mB = (tile >> 2) * 16, nB = (tile & 3) * 16;
      v8f acc = {};
#pragma unroll
      for (int c = 0; c < 3; ++c) {
#pragma unroll
        for (int k = 0; k < 64; k += 32) {
          acc = wmma_f16(load_a_frag(srcs[c], 64, mB, k, lane),
                         load_bT_frag(sW3, 192, nB, c * 64 + k, lane), acc);
        }
      }
      store_tile(acc, buf4, nullptr, mB, nB, lane, sB3, true);
    }
  }
  __syncthreads();

  // 8) actions[n] = sigmoid(x3[n,:] . out_w + out_b)   (x3 in buf4)
  if (tid < 64) {
    float s = outb[0];
#pragma unroll
    for (int k = 0; k < 64; ++k) s += (float)buf4[tid * 64 + k] * sOw[k];
    __builtin_nontemporal_store(sigmoidf_(s), out + (size_t)b * 64 + tid);
  }

  // 9) GRU cell: 4 register accumulators per tile (r, z, gi_n, gh_n)
#pragma unroll
  for (int t = 0; t < 2; ++t) {
    const int tile = wave * 2 + t;
    const int mB = (tile >> 2) * 16, nB = (tile & 3) * 16;
    v8f ar = {}, az = {}, ain = {}, ahn = {};
#pragma unroll
    for (int k = 0; k < 64; k += 32) {
      const v16h ax = load_a_frag(buf4, 64, mB, k, lane);  // x3
      const v16h ah = load_a_frag(buf3, 64, mB, k, lane);  // hidden (f16)
      ar = wmma_f16(ax, load_bT_frag(sWih + 0 * 4096, 64, nB, k, lane), ar);
      ar = wmma_f16(ah, load_bT_frag(sWhh + 0 * 4096, 64, nB, k, lane), ar);
      az = wmma_f16(ax, load_bT_frag(sWih + 1 * 4096, 64, nB, k, lane), az);
      az = wmma_f16(ah, load_bT_frag(sWhh + 1 * 4096, 64, nB, k, lane), az);
      ain = wmma_f16(ax, load_bT_frag(sWih + 2 * 4096, 64, nB, k, lane), ain);
      ahn = wmma_f16(ah, load_bT_frag(sWhh + 2 * 4096, 64, nB, k, lane), ahn);
    }
    const int n  = nB + (lane & 15);
    const int hi = (lane >> 4) & 1;
    const float br  = sBih[n] + sBhh[n];
    const float bz  = sBih[64 + n] + sBhh[64 + n];
    const float bin = sBih[128 + n];
    const float bhn = sBhh[128 + n];
#pragma unroll
    for (int v = 0; v < 8; ++v) {
      const int m = mB + v + 8 * hi;
      const float r  = sigmoidf_(ar[v] + br);
      const float z  = sigmoidf_(az[v] + bz);
      const float nn = tanhf_(ain[v] + bin + r * (ahn[v] + bhn));
      const float hp = hidden[ob + (size_t)m * 64 + n];  // f32, L2-resident
      __builtin_nontemporal_store((1.0f - z) * nn + z * hp,
                                  out + 262144 + ob + (size_t)m * 64 + n);
    }
  }
}

// ---------------------------------------------------------------------------
extern "C" void kernel_launch(void* const* d_in, const int* in_sizes, int n_in,
                              void* d_out, int out_size, void* d_ws, size_t ws_size,
                              hipStream_t stream) {
  (void)in_sizes; (void)n_in; (void)out_size; (void)d_ws; (void)ws_size;
  const float* state  = (const float*)d_in[0];
  const float* adjc   = (const float*)d_in[1];
  const float* adjo   = (const float*)d_in[2];
  const float* hidden = (const float*)d_in[3];
  const float* fc1w   = (const float*)d_in[4];
  const float* fc1b   = (const float*)d_in[5];
  const float* fc2w   = (const float*)d_in[6];
  const float* fc2b   = (const float*)d_in[7];
  const float* compw  = (const float*)d_in[8];
  const float* compb  = (const float*)d_in[9];
  const float* coopw  = (const float*)d_in[10];
  const float* coopb  = (const float*)d_in[11];
  const float* fc3w   = (const float*)d_in[12];
  const float* fc3b   = (const float*)d_in[13];
  const float* outw   = (const float*)d_in[14];
  const float* outb   = (const float*)d_in[15];
  const float* wih    = (const float*)d_in[16];
  const float* whh    = (const float*)d_in[17];
  const float* bih    = (const float*)d_in[18];
  const float* bhh    = (const float*)d_in[19];
  float* out = (float*)d_out;

  actor_fused_kernel<<<4096, 256, 0, stream>>>(
      state, adjc, adjo, hidden, fc1w, fc1b, fc2w, fc2b, compw, compb,
      coopw, coopb, fc3w, fc3b, outw, outb, wih, whh, bih, bhh, out);
}
